// Self_Attention_3D_28887950032994
// MI455X (gfx1250) — matrix operand order repaired and checked by reading, not compiled, not run on hardware
//
#include <hip/hip_runtime.h>
#include <hip/hip_bf16.h>
#include <stdint.h>

// ---- CDNA5 WMMA types -------------------------------------------------------
typedef __attribute__((ext_vector_type(16))) __bf16 v16bf;
typedef __attribute__((ext_vector_type(8)))  float  v8f;
typedef __attribute__((ext_vector_type(2)))  __bf16 bf16x2;
typedef __attribute__((ext_vector_type(2)))  float  f32x2;

union Frag16 {               // 16 bf16 elements == 8 VGPRs
    v16bf v;
    uint4 q[2];
    unsigned short s[16];
};

__device__ __forceinline__ v8f wmma_bf16(v16bf a, v16bf b, v8f c) {
    // D = A(16x32 bf16) * B(32x16 bf16) + C(16x16 f32)
    return __builtin_amdgcn_wmma_f32_16x16x32_bf16(
        /*neg_a=*/false, a, /*neg_b=*/false, b,
        /*c_mod=*/(short)0, c, /*reuse_a=*/false, /*reuse_b=*/false);
}

// ---- f32 -> bf16: let the backend pick v_cvt_pk_bf16_f32 when available ----
__device__ __forceinline__ unsigned pack_bf16(float a, float b) {
    f32x2 f = {a, b};
    bf16x2 r = __builtin_convertvector(f, bf16x2);
    unsigned u;
    __builtin_memcpy(&u, &r, 4);
    return u;
}
__device__ __forceinline__ unsigned short f2bf1(float a) {
    __bf16 r = (__bf16)a;
    unsigned short s;
    __builtin_memcpy(&s, &r, 2);
    return s;
}

// exp2 (scores are pre-scaled by log2(e) in the Q projection)
#if __has_builtin(__builtin_amdgcn_exp2f)
#define EXP2F(x) __builtin_amdgcn_exp2f(x)
#else
#define EXP2F(x) exp2f(x)
#endif

__device__ __forceinline__ float redmax16(float v) {
    v = fmaxf(v, __shfl_xor(v, 1, 32));
    v = fmaxf(v, __shfl_xor(v, 2, 32));
    v = fmaxf(v, __shfl_xor(v, 4, 32));
    v = fmaxf(v, __shfl_xor(v, 8, 32));
    return v;
}
__device__ __forceinline__ float redsum16(float v) {
    v += __shfl_xor(v, 1, 32);
    v += __shfl_xor(v, 2, 32);
    v += __shfl_xor(v, 4, 32);
    v += __shfl_xor(v, 8, 32);
    return v;
}

static constexpr int   C     = 256;
static constexpr int   CQ    = 32;
static constexpr int   N     = 4096;
static constexpr float LOG2E = 1.44269504088896340736f;

// ============================================================================
// Kernel 0: one-time f32 -> bf16 conversion of the three weight matrices.
// Wq is pre-scaled by log2(e) so attention scores land in the exp2 domain.
// ============================================================================
__global__ __launch_bounds__(256) void prep_weights_kernel(
    const float* __restrict__ Wq, const float* __restrict__ Wk,
    const float* __restrict__ Wv,
    unsigned* __restrict__ Wqb, unsigned* __restrict__ Wkb,
    unsigned* __restrict__ Wvb)
{
    const int i = blockIdx.x * blockDim.x + threadIdx.x;   // pair index
    if (i < 4096) {
        Wqb[i] = pack_bf16(Wq[2 * i] * LOG2E, Wq[2 * i + 1] * LOG2E);
        Wkb[i] = pack_bf16(Wk[2 * i], Wk[2 * i + 1]);
    }
    if (i < 32768)
        Wvb[i] = pack_bf16(Wv[2 * i], Wv[2 * i + 1]);
}

// ============================================================================
// Kernel 1: fused QKV 1x1-conv projections (bf16 WMMA, f32 accumulate)
// grid = B * (N/16) blocks, 256 threads (8 waves)
// outputs: qT [B][N][32] bf16 (scaled by log2e), kT [B][N][32] bf16,
//          vbf [B][C][N] bf16
// ============================================================================
__global__ __launch_bounds__(256) void qkv_proj_kernel(
    const float* __restrict__ x,
    const unsigned short* __restrict__ Wqb, const float* __restrict__ bq,
    const unsigned short* __restrict__ Wkb, const float* __restrict__ bk,
    const unsigned short* __restrict__ Wvb, const float* __restrict__ bv,
    unsigned short* __restrict__ qT, unsigned short* __restrict__ kT,
    unsigned short* __restrict__ vbf)
{
    __shared__ __align__(16) unsigned short xT[16 * C];   // [n][c] bf16, 8KB

    const int bid = blockIdx.x;
    const int b   = bid >> 8;
    const int n0  = (bid & 255) << 4;

    // ---- stage x tile [C x 16] -> LDS transposed bf16 [16 x C] -------------
    {
        const int c = threadIdx.x;                         // 0..255
        const float* xp = x + ((size_t)b * C + c) * N + n0;
        float4 f[4];
        const float4* xp4 = (const float4*)xp;
        f[0] = xp4[0]; f[1] = xp4[1]; f[2] = xp4[2]; f[3] = xp4[3];
        const float* ff = (const float*)f;
        #pragma unroll
        for (int n = 0; n < 16; ++n)
            xT[n * C + c] = f2bf1(ff[n]);
    }
    __syncthreads();

    const int wave = threadIdx.x >> 5;
    const int lane = threadIdx.x & 31;
    const int nl   = lane & 15;
    const int h    = lane >> 4;

    // 20 output tiles: 0..1 -> q, 2..3 -> k, 4..19 -> v
    for (int t = wave; t < 20; t += 8) {
        const unsigned short* Wp; const float* bp; int o0; int kind;
        if (t < 2)      { Wp = Wqb; bp = bq; o0 = t * 16;        kind = 0; }
        else if (t < 4) { Wp = Wkb; bp = bk; o0 = (t - 2) * 16;  kind = 1; }
        else            { Wp = Wvb; bp = bv; o0 = (t - 4) * 16;  kind = 2; }

        v8f acc = {};
        #pragma unroll
        for (int ks = 0; ks < 8; ++ks) {
            // A fragment (bf16, direct b128 loads): row o0+nl, K runs of 8
            const unsigned short* wrow = Wp + (size_t)(o0 + nl) * C + ks * 32;
            Frag16 a;
            a.q[0] = *(const uint4*)(wrow + 16 * h);
            a.q[1] = *(const uint4*)(wrow + 16 * h + 8);
            // B fragment from LDS: column nl, K = ks*32 + 16h + e (contiguous)
            Frag16 bm;
            const uint4* lp = (const uint4*)&xT[nl * C + ks * 32 + 16 * h];
            bm.q[0] = lp[0]; bm.q[1] = lp[1];

            acc = wmma_bf16(a.v, bm.v, acc);
        }

        // bias add (rows m = r + 8h); q bias carries the log2e score scaling
        const float bscale = (kind == 0) ? LOG2E : 1.0f;
        float4 bv0 = *(const float4*)(bp + o0 + 8 * h);
        float4 bv1 = *(const float4*)(bp + o0 + 8 * h + 4);
        float bias[8] = { bv0.x, bv0.y, bv0.z, bv0.w, bv1.x, bv1.y, bv1.z, bv1.w };
        #pragma unroll
        for (int r = 0; r < 8; ++r) acc[r] += bias[r] * bscale;

        if (kind == 2) {
            // v stored [C][N]: scattered bf16 stores (row per r)
            #pragma unroll
            for (int r = 0; r < 8; ++r)
                vbf[((size_t)b * C + o0 + 8 * h + r) * N + n0 + nl] = f2bf1(acc[r]);
        } else {
            // q/k stored transposed [N][32]: 8 contiguous bf16 -> one b128
            unsigned pk[4] __attribute__((aligned(16)));
            #pragma unroll
            for (int r = 0; r < 4; ++r)
                pk[r] = pack_bf16(acc[2 * r], acc[2 * r + 1]);
            unsigned short* dst = (kind == 0 ? qT : kT)
                + ((size_t)b * N + n0 + nl) * CQ + o0 + 8 * h;
            *(uint4*)dst = *(const uint4*)pk;
        }
    }
}

// ============================================================================
// Kernel 2: flash attention + residual.
// grid = B*(N/16) blocks of 64 threads (2 waves). Both waves share the block's
// 16-row i-tile; wave w owns channels [128w, 128w+128) as 8 c-tile f32
// accumulators. j-tile = 64: per wave per iter 4 S-WMMA + 16 PV-WMMA.
// Scores are in the exp2 domain (q pre-scaled by log2e) -> bare v_exp_f32.
// Barrier-free: per-wave private LDS P buffer; l-reduction deferred to end.
// ============================================================================
__global__ __launch_bounds__(64) void attn_kernel(
    const float* __restrict__ x, const float* __restrict__ gamma,
    const unsigned short* __restrict__ qT, const unsigned short* __restrict__ kT,
    const unsigned short* __restrict__ vbf, float* __restrict__ out)
{
    __shared__ __align__(16) unsigned short p_lds[2 * 16 * 64]; // 2 waves x (16x64) bf16

    const int bid = blockIdx.x;
    const int b   = bid >> 8;
    const int i0  = (bid & 255) << 4;

    const int wave = threadIdx.x >> 5;
    const int lane = threadIdx.x & 31;
    const int nl   = lane & 15;
    const int h    = lane >> 4;
    unsigned short* pw = p_lds + wave * (16 * 64);
    const int cbase = wave * 128;            // this wave's channel range

    // q A-fragment (held for the whole loop): row i0+nl, K runs of 8
    Frag16 qa;
    {
        const unsigned short* qrow = qT + ((size_t)b * N + i0 + nl) * CQ;
        qa.q[0] = *(const uint4*)(qrow + 16 * h);
        qa.q[1] = *(const uint4*)(qrow + 16 * h + 8);
    }

    v8f O[8];
    #pragma unroll
    for (int ct = 0; ct < 8; ++ct) O[ct] = (v8f){};
    float mrow[8], lpart[8];
    #pragma unroll
    for (int r = 0; r < 8; ++r) { mrow[r] = -1e30f; lpart[r] = 0.0f; }

    const unsigned short* kB = kT + (size_t)b * N * CQ;
    const unsigned short* vB = vbf + (size_t)b * C * N;

    for (int j0 = 0; j0 < N; j0 += 64) {
        // ---- S = q . k^T for 64 j columns (4 WMMA) -------------------------
        v8f s[4];
        #pragma unroll
        for (int jt = 0; jt < 4; ++jt) {
            Frag16 kf;
            const uint4* kp =
                (const uint4*)(kB + (size_t)(j0 + jt * 16 + nl) * CQ + 16 * h);
            kf.q[0] = kp[0]; kf.q[1] = kp[1];
            v8f z = {};
            s[jt] = wmma_bf16(qa.v, kf.v, z);
        }

        // ---- online softmax in exp2 domain (rows in 16-lane halves) --------
        v8f scv;
        #pragma unroll
        for (int r = 0; r < 8; ++r) {
            float tmax = fmaxf(fmaxf(s[0][r], s[1][r]), fmaxf(s[2][r], s[3][r]));
            tmax = redmax16(tmax);
            float mn = fmaxf(mrow[r], tmax);
            float sc = EXP2F(mrow[r] - mn);
            mrow[r] = mn;
            float p0 = EXP2F(s[0][r] - mn);
            float p1 = EXP2F(s[1][r] - mn);
            float p2 = EXP2F(s[2][r] - mn);
            float p3 = EXP2F(s[3][r] - mn);
            // per-lane partial sum; cross-lane reduction deferred to the end
            lpart[r] = lpart[r] * sc + ((p0 + p1) + (p2 + p3));
            scv[r] = sc;
            unsigned short* prow = pw + (r + 8 * h) * 64 + nl;
            prow[0]  = f2bf1(p0);
            prow[16] = f2bf1(p1);
            prow[32] = f2bf1(p2);
            prow[48] = f2bf1(p3);
        }
        // vectorized accumulator rescale (v_pk_mul_f32-friendly)
        #pragma unroll
        for (int ct = 0; ct < 8; ++ct) O[ct] *= scv;

        // ---- reload P as two A fragments (bf16 16x32 each) -----------------
        Frag16 pa0, pa1;
        {
            const unsigned short* prow = pw + nl * 64;
            pa0.q[0] = *(const uint4*)(prow + 16 * h);
            pa0.q[1] = *(const uint4*)(prow + 8 + 16 * h);
            pa1.q[0] = *(const uint4*)(prow + 32 + 16 * h);
            pa1.q[1] = *(const uint4*)(prow + 40 + 16 * h);
        }

        // ---- O[ct] += P x V^T for this wave's eight 16-channel tiles -------
        #pragma unroll
        for (int ct = 0; ct < 8; ++ct) {
            const unsigned short* vr =
                vB + (size_t)(cbase + ct * 16 + nl) * N + j0 + 16 * h;
            Frag16 v0, v1;
            v0.q[0] = ((const uint4*)vr)[0];        v0.q[1] = ((const uint4*)vr)[1];
            v1.q[0] = ((const uint4*)(vr + 32))[0]; v1.q[1] = ((const uint4*)(vr + 32))[1];
            O[ct] = wmma_bf16(pa0.v, v0.v, O[ct]);
            O[ct] = wmma_bf16(pa1.v, v1.v, O[ct]);
        }
    }

    // ---- finalize: y = gamma * O/l + x, contiguous float4 stores -----------
    const float g = gamma[0];
    float lrow[8];
    #pragma unroll
    for (int r = 0; r < 8; ++r) lrow[r] = 1.0f / redsum16(lpart[r]);

    #pragma unroll
    for (int ct = 0; ct < 8; ++ct) {
        const int c = cbase + ct * 16 + nl;
        const size_t base = ((size_t)b * C + c) * N + i0 + 8 * h;
        float4 x0 = *(const float4*)(x + base);
        float4 x1 = *(const float4*)(x + base + 4);
        const float* xf0 = (const float*)&x0;
        const float* xf1 = (const float*)&x1;
        float res[8] __attribute__((aligned(16)));
        #pragma unroll
        for (int r = 0; r < 8; ++r) {
            float xr = (r < 4) ? xf0[r] : xf1[r - 4];
            res[r] = g * (O[ct][r] * lrow[r]) + xr;
        }
        *(float4*)(out + base)     = *(const float4*)res;
        *(float4*)(out + base + 4) = *(const float4*)(res + 4);
    }
}

// ============================================================================
extern "C" void kernel_launch(void* const* d_in, const int* in_sizes, int n_in,
                              void* d_out, int out_size, void* d_ws, size_t ws_size,
                              hipStream_t stream) {
    const float* x     = (const float*)d_in[0];
    const float* Wq    = (const float*)d_in[1];
    const float* bq    = (const float*)d_in[2];
    const float* Wk    = (const float*)d_in[3];
    const float* bk    = (const float*)d_in[4];
    const float* Wv    = (const float*)d_in[5];
    const float* bv    = (const float*)d_in[6];
    const float* gamma = (const float*)d_in[7];
    float* out = (float*)d_out;

    constexpr int B = 4;
    unsigned short* qT  = (unsigned short*)d_ws;                 // 4*4096*32 bf16 = 1 MB
    unsigned short* kT  = qT + (size_t)B * N * CQ;               // 1 MB
    unsigned short* vbf = kT + (size_t)B * N * CQ;               // 4*256*4096 bf16 = 8 MB
    unsigned short* Wqb = vbf + (size_t)B * C * N;               // 16 KB
    unsigned short* Wkb = Wqb + CQ * C;                          // 16 KB
    unsigned short* Wvb = Wkb + CQ * C;                          // 128 KB

    hipLaunchKernelGGL(prep_weights_kernel, dim3(128), dim3(256), 0, stream,
                       Wq, Wk, Wv, (unsigned*)Wqb, (unsigned*)Wkb, (unsigned*)Wvb);

    dim3 grid(B * (N / 16));
    hipLaunchKernelGGL(qkv_proj_kernel, grid, dim3(256), 0, stream,
                       x, Wqb, bq, Wkb, bk, Wvb, bv, qT, kT, vbf);
    hipLaunchKernelGGL(attn_kernel, grid, dim3(64), 0, stream,
                       x, gamma, qT, kT, vbf, out);
    (void)in_sizes; (void)n_in; (void)out_size; (void)ws_size;
}
